// RIOS_H_87368224735873
// MI455X (gfx1250) — matrix operand order repaired and checked by reading, not compiled, not run on hardware
//
#include <hip/hip_runtime.h>
#include <hip/hip_bf16.h>
#include <stdint.h>

// ---------------- constants ----------------
#define kH       128
#define kT       2048
#define kB       4096
#define kWaves   4                 // waves per workgroup (1 WG per WGP: LDS-bound)
#define kThreads (kWaves * 32)
#define kRowsWG  (kWaves * 16)
#define kPitch   132               // bf16 elems per LDS row (128 + pad, 8B aligned, bank-skewed)

typedef __bf16 bf16_t;
typedef __bf16 v16bf __attribute__((ext_vector_type(16)));
typedef float  v8f   __attribute__((ext_vector_type(8)));

union FragU { v16bf v; unsigned long long q[4]; };

// smem elems: 2*512*kPitch bf16 weights + kWaves*16*kPitch bf16 h1 + (6*512 + kWaves*16) f32
#define kSmemBytes ((size_t)(2 * 512 * kPitch + kWaves * 16 * kPitch) * 2 + \
                    (size_t)(6 * 512 + kWaves * 16) * 4)

__device__ __forceinline__ float sigm(float x) {
  return __builtin_amdgcn_rcpf(1.0f + __expf(-x));
}
__device__ __forceinline__ float ftanh(float x) {
  x = fminf(fmaxf(x, -15.0f), 15.0f);
  float e = __expf(2.0f * x);
  return (e - 1.0f) * __builtin_amdgcn_rcpf(e + 1.0f);
}

// A fragment: 16x32 bf16, M = lane&15 (row pointer passed in), two 8-elem runs.
// lanes 0-15: K = base+0..7 (V0-3), base+16..23 (V4-7); lanes 16-31: +8.
__device__ __forceinline__ v16bf load_afrag(const bf16_t* row, int kt, int hi) {
  const bf16_t* p = row + kt * 32 + hi * 8;
  FragU f;
  f.q[0] = *(const unsigned long long*)(p + 0);
  f.q[1] = *(const unsigned long long*)(p + 4);
  f.q[2] = *(const unsigned long long*)(p + 16);
  f.q[3] = *(const unsigned long long*)(p + 20);
  return f.v;
}
// B fragment: 32x16 bf16 from Whh stored [n][k] (B = Whh^T). N = lane&15,
// lanes 0-15 hold K=base+0..7, base+16..23; lanes 16-31: +8 (same as A).
__device__ __forceinline__ v16bf load_bfrag(const bf16_t* w, int n, int kt, int hi) {
  const bf16_t* p = w + n * kPitch + kt * 32 + hi * 8;
  FragU f;
  f.q[0] = *(const unsigned long long*)(p + 0);
  f.q[1] = *(const unsigned long long*)(p + 4);
  f.q[2] = *(const unsigned long long*)(p + 16);
  f.q[3] = *(const unsigned long long*)(p + 20);
  return f.v;
}

// Intra-wave LDS RAW: DS ops from one wave execute in order in the LDS pipeline,
// so no counter drain is required -- only compiler ordering.
#define LDS_ORDER() asm volatile("" ::: "memory")

// ---------------- kernel 1: inverse mask-sum per timestep + zero loss ----------------
__global__ void rios_den(const float* __restrict__ masks, float* __restrict__ inv_den,
                         float* __restrict__ out) {
  __shared__ float red[256];
  int t = blockIdx.x;
  float s = 0.0f;
  for (int b = threadIdx.x; b < kB; b += 256) s += masks[(size_t)b * kT + t];
  red[threadIdx.x] = s;
  __syncthreads();
  for (int o = 128; o > 0; o >>= 1) {
    if (threadIdx.x < o) red[threadIdx.x] += red[threadIdx.x + o];
    __syncthreads();
  }
  if (threadIdx.x == 0) inv_den[t] = 1.0f / (red[0] + 1e-5f);
  if (blockIdx.x == 0 && threadIdx.x == 0) out[0] = 0.0f;
}

// ---------------- kernel 2: passthrough outputs ----------------
__global__ void rios_copy(const float* __restrict__ values, const float* __restrict__ masks,
                          const float* __restrict__ evalm, float* __restrict__ out) {
  size_t i = (size_t)blockIdx.x * blockDim.x + threadIdx.x;
  const size_t N = (size_t)kB * kT;
  if (i < N) {
    out[1 + N + i]     = evalm[i];          // eval_masks
    out[1 + 2 * N + i] = values[3 * i + 2]; // evals = values[:,:,2]
    out[1 + 3 * N + i] = masks[i];          // masks
  }
}

// ---------------- kernel 3: main recurrent kernel ----------------
__global__ void __launch_bounds__(kThreads, 1)
rios_main(const float* __restrict__ values, const float* __restrict__ masks,
          const float* __restrict__ deltas,
          const float* __restrict__ Wih0, const float* __restrict__ Whh0,
          const float* __restrict__ bih0, const float* __restrict__ bhh0,
          const float* __restrict__ Wih1, const float* __restrict__ Whh1,
          const float* __restrict__ bih1, const float* __restrict__ bhh1,
          const float* __restrict__ td_W, const float* __restrict__ td_b,
          const float* __restrict__ hr_W, const float* __restrict__ hr_b,
          const float* __restrict__ inv_den, float* __restrict__ out) {
  extern __shared__ char smem[];
  bf16_t* whh1s = (bf16_t*)smem;                       // [512][kPitch]
  bf16_t* whh0s = whh1s + 512 * kPitch;                // [512][kPitch]
  bf16_t* h1s   = whh0s + 512 * kPitch;                // [kWaves][16][kPitch]
  float*  bias1 = (float*)(h1s + kWaves * 16 * kPitch);
  float*  bias0 = bias1 + 512;
  float*  w1a   = bias0 + 512;
  float*  w1b   = w1a + 512;
  float*  w0a   = w1b + 512;
  float*  w0b   = w0a + 512;
  float*  xhbuf = w0b + 512;                           // [kWaves][16]

  const int tid = threadIdx.x;

  // ---- one-time weight prep: fp32 -> bf16 into padded LDS, combined biases ----
  for (int idx = tid; idx < 512 * kH; idx += kThreads) {
    int n = idx >> 7, k = idx & 127;
    whh1s[n * kPitch + k] = (bf16_t)Whh1[idx];
    whh0s[n * kPitch + k] = (bf16_t)Whh0[idx];
  }
  for (int n = tid; n < 512; n += kThreads) {
    bias1[n] = bih1[n] + bhh1[n];
    bias0[n] = bih0[n] + bhh0[n];
    w1a[n] = Wih1[2 * n]; w1b[n] = Wih1[2 * n + 1];
    w0a[n] = Wih0[2 * n]; w0b[n] = Wih0[2 * n + 1];
  }
  for (int i = tid; i < kWaves * 16 * kPitch; i += kThreads) h1s[i] = (bf16_t)0.0f;
  __syncthreads();

  const int lane = tid & 31;
  const int wave = tid >> 5;
  const int col  = lane & 15;     // N-column / local batch row for scalar path
  const int hi   = lane >> 4;     // lane half
  const int hib  = hi * 8;

  bf16_t* myh1  = h1s + wave * 16 * kPitch;
  bf16_t* myrow = myh1 + col * kPitch;     // A fragment row (M = col)
  float*  myxh  = xhbuf + wave * 16;

  const int row = blockIdx.x * kRowsWG + wave * 16 + col;  // global batch row

  // per-lane constant preloads (column-indexed small weights)
  float tdWl[8], tdbl[8], hrWl[8];
#pragma unroll
  for (int nc = 0; nc < 8; nc++) {
    tdWl[nc] = td_W[nc * 16 + col];
    tdbl[nc] = td_b[nc * 16 + col];
    hrWl[nc] = hr_W[nc * 16 + col];
  }
  const float hrb = hr_b[0];

  v8f c1[8];
#pragma unroll
  for (int nc = 0; nc < 8; nc++)
#pragma unroll
    for (int r = 0; r < 8; r++) c1[nc][r] = 0.0f;
  float xhp[8];
#pragma unroll
  for (int r = 0; r < 8; r++) xhp[r] = 0.0f;
  float loss_acc = 0.0f;

  const float* vrow = values + (size_t)row * kT * 3;
  const float* mrow = masks + (size_t)row * kT;
  const float* drow = deltas + (size_t)row * kT * 3;
  float* imp = out + 1;

  // preload t = 0 scalars
  float cxy = vrow[2], ca = vrow[0], cb = vrow[1], cm = mrow[0], cd = drow[2];

  for (int t = 0; t < kT; ++t) {
    // software pipeline: issue next-step input loads now
    float nxy = 0.f, na = 0.f, nb = 0.f, nm = 0.f, nd = 0.f;
    if (t + 1 < kT) {
      const float* vp = vrow + (size_t)(t + 1) * 3;
      nxy = vp[2]; na = vp[0]; nb = vp[1];
      nm = mrow[t + 1];
      nd = drow[(size_t)(t + 1) * 3 + 2];
    }
    const float invd = inv_den[t];

    // ---- x_h = h0 . hr_W + hr_b  (xhp holds per-lane partials from last step) ----
    float xh8[8];
#pragma unroll
    for (int r = 0; r < 8; r++) {
      float s = xhp[r];
      s += __shfl_xor(s, 1, 32);
      s += __shfl_xor(s, 2, 32);
      s += __shfl_xor(s, 4, 32);
      s += __shfl_xor(s, 8, 32);
      xh8[r] = s + hrb;   // x_h for row hib+r, broadcast within half
      xhp[r] = 0.0f;
    }
    if (col == 0) {
#pragma unroll
      for (int r = 0; r < 8; r++) myxh[hib + r] = xh8[r];
    }
    LDS_ORDER();                       // intra-wave DS ops are in-order: no drain needed
    const float xh = myxh[col];        // x_h for this lane's scalar row
    LDS_ORDER();

    const float xc = cm * cxy + (1.0f - cm) * xh;
    const float dd = cxy - xh;
    const float lp = (hi == 0) ? dd * dd * cm : 0.0f;
    if (hi == 0) imp[(size_t)row * kT + t] = xc;   // imputation output

    // row-broadcasts into C/D layout (element r <-> batch row hib+r)
    float d8[8], m8[8], xc8[8], a8[8], b8[8];
#pragma unroll
    for (int r = 0; r < 8; r++) {
      const int src = hib + r;
      d8[r]  = __shfl(cd, src, 32);
      m8[r]  = __shfl(cm, src, 32);
      xc8[r] = __shfl(xc, src, 32);
      a8[r]  = __shfl(ca, src, 32);
      b8[r]  = __shfl(cb, src, 32);
    }

    // ---- LSTM1: g = in1 @ Wih1^T + b + h1 @ Whh1^T ----
    v16bf af[4];
#pragma unroll
    for (int kt = 0; kt < 4; kt++) af[kt] = load_afrag(myrow, kt, hi);

#pragma unroll
    for (int nc = 0; nc < 8; nc++) {
      v8f acc[4];
#pragma unroll
      for (int G = 0; G < 4; G++) {
        const int n = G * 128 + nc * 16 + col;
        const float bb = bias1[n], wa = w1a[n], wb = w1b[n];
#pragma unroll
        for (int r = 0; r < 8; r++) acc[G][r] = bb + a8[r] * wa + b8[r] * wb;
      }
#pragma unroll
      for (int kt = 0; kt < 4; kt++) {
#pragma unroll
        for (int G = 0; G < 4; G++) {
          v16bf bf = load_bfrag(whh1s, G * 128 + nc * 16 + col, kt, hi);
          acc[G] = __builtin_amdgcn_wmma_f32_16x16x32_bf16(
              false, af[kt], false, bf, (short)0, acc[G], false, false);
        }
      }
#pragma unroll
      for (int r = 0; r < 8; r++) {
        const float iv = sigm(acc[0][r]);
        const float fv = sigm(acc[1][r]);
        const float gv = ftanh(acc[2][r]);
        const float ov = sigm(acc[3][r]);
        const float c  = fv * c1[nc][r] + iv * gv;
        c1[nc][r] = c;
        const float gam = __expf(-fmaxf(d8[r] * tdWl[nc] + tdbl[nc], 0.0f));
        const float hn  = ov * ftanh(c) * gam;   // h1n = h1n * gamma_h (carried state)
        myh1[(hib + r) * kPitch + nc * 16 + col] = (bf16_t)hn;
      }
    }
    LDS_ORDER();   // stores above precede loads below in the in-order DS pipe

    // ---- LSTM0: hidden input is h1n (just written); cell input is c1n; only h0 kept ----
#pragma unroll
    for (int kt = 0; kt < 4; kt++) af[kt] = load_afrag(myrow, kt, hi);

#pragma unroll
    for (int nc = 0; nc < 8; nc++) {
      v8f acc[4];
#pragma unroll
      for (int G = 0; G < 4; G++) {
        const int n = G * 128 + nc * 16 + col;
        const float bb = bias0[n], wa = w0a[n], wb = w0b[n];
#pragma unroll
        for (int r = 0; r < 8; r++) acc[G][r] = bb + xc8[r] * wa + m8[r] * wb;
      }
#pragma unroll
      for (int kt = 0; kt < 4; kt++) {
#pragma unroll
        for (int G = 0; G < 4; G++) {
          v16bf bf = load_bfrag(whh0s, G * 128 + nc * 16 + col, kt, hi);
          acc[G] = __builtin_amdgcn_wmma_f32_16x16x32_bf16(
              false, af[kt], false, bf, (short)0, acc[G], false, false);
        }
      }
#pragma unroll
      for (int r = 0; r < 8; r++) {
        const float iv = sigm(acc[0][r]);
        const float fv = sigm(acc[1][r]);
        const float gv = ftanh(acc[2][r]);
        const float ov = sigm(acc[3][r]);
        const float c2 = fv * c1[nc][r] + iv * gv;     // c discarded afterwards
        const float h0v = ov * ftanh(c2);
        xhp[r] += h0v * hrWl[nc];                      // fold h0 straight into next x_h dot
      }
    }

    loss_acc += lp * invd;

    cxy = nxy; ca = na; cb = nb; cm = nm; cd = nd;
  }

  // ---- loss: reduce wave partials, one atomic per wave ----
  float ls = loss_acc;
  ls += __shfl_xor(ls, 1, 32);
  ls += __shfl_xor(ls, 2, 32);
  ls += __shfl_xor(ls, 4, 32);
  ls += __shfl_xor(ls, 8, 32);
  ls += __shfl_xor(ls, 16, 32);
  if (lane == 0) atomicAdd(out, ls);
}

// ---------------- host entry ----------------
extern "C" void kernel_launch(void* const* d_in, const int* in_sizes, int n_in,
                              void* d_out, int out_size, void* d_ws, size_t ws_size,
                              hipStream_t stream) {
  const float* values = (const float*)d_in[0];
  const float* masks  = (const float*)d_in[1];
  const float* deltas = (const float*)d_in[2];
  const float* evalm  = (const float*)d_in[3];
  const float* Wih0   = (const float*)d_in[4];
  const float* Whh0   = (const float*)d_in[5];
  const float* bih0   = (const float*)d_in[6];
  const float* bhh0   = (const float*)d_in[7];
  const float* Wih1   = (const float*)d_in[8];
  const float* Whh1   = (const float*)d_in[9];
  const float* bih1   = (const float*)d_in[10];
  const float* bhh1   = (const float*)d_in[11];
  const float* td_W   = (const float*)d_in[12];
  const float* td_b   = (const float*)d_in[13];
  const float* hr_W   = (const float*)d_in[14];
  const float* hr_b   = (const float*)d_in[15];
  float* out = (float*)d_out;
  float* inv_den = (float*)d_ws;   // 2048 floats of scratch

  rios_den<<<kT, 256, 0, stream>>>(masks, inv_den, out);

  const size_t N = (size_t)kB * kT;
  const int cblocks = (int)((N + 255) / 256);
  rios_copy<<<cblocks, 256, 0, stream>>>(values, masks, evalm, out);

  rios_main<<<kB / kRowsWG, kThreads, kSmemBytes, stream>>>(
      values, masks, deltas, Wih0, Whh0, bih0, bhh0, Wih1, Whh1, bih1, bhh1,
      td_W, td_b, hr_W, hr_b, inv_den, out);
}
